// Circulant_51685636440286
// MI455X (gfx1250) — matrix operand order repaired and checked by reading, not compiled, not run on hardware
//
#include <hip/hip_runtime.h>

// Problem constants (from the reference)
#define NDIM 4096      // in_channels / circulant size
#define BDIM 8192      // batch rows
// Tiling: block tile 128(M) x 256(N), K-step 32; 8 wave32, each 64x64 (4x4 frags)
#define BM 128
#define BN 256
#define BK 32
#define MF 4
#define NF 4
#define XP 40                 // ushort pitch for staged X rows (80 B, 16B aligned)
#define WLEN (2 * NDIM + 24)  // duplicated-w length in ushorts (covers s+16 overreach)

typedef __attribute__((ext_vector_type(16))) __bf16          v16bf;
typedef __attribute__((ext_vector_type(8)))  float           v8f;
typedef __attribute__((ext_vector_type(8)))  unsigned short  ushort8;

union Frag32 {
    v16bf          v;
    ushort8        h[2];
    unsigned int   u[8];
};

#if __has_builtin(__builtin_amdgcn_alignbit)
#define ALIGNBIT(hi, lo, s) __builtin_amdgcn_alignbit((hi), (lo), (s))
#else
#define ALIGNBIT(hi, lo, s) \
    ((unsigned int)((((unsigned long long)(hi) << 32) | (unsigned long long)(lo)) >> (s)))
#endif

// v_perm selector: result = (src_a[31:16] << 16) | src_b[31:16]
// bytes of {a,b}: b=bytes0..3, a=bytes4..7; pick 2,3 (b hi16) then 6,7 (a hi16)
#define PERM_HI16 0x07060302u

__global__ __launch_bounds__(256)
void circulant_wmma(const float* __restrict__ x, const float* __restrict__ w,
                    float* __restrict__ out) {
    // w duplicated (index space 0..2N) and split into bf16 hi/lo (truncation split)
    __shared__ alignas(16) unsigned short sWhi[WLEN];
    __shared__ alignas(16) unsigned short sWlo[WLEN];
    // staged X tile, split bf16, row pitch XP ushorts (16B-aligned rows)
    __shared__ alignas(16) unsigned short sXhi[BM * XP];
    __shared__ alignas(16) unsigned short sXlo[BM * XP];

    const int tid  = threadIdx.x;
    const int lane = tid & 31;
    const int wid  = tid >> 5;      // 0..7 (8 wave32)
    const int wm   = wid & 1;       // wave-row: 2 x 64 rows
    const int wn   = wid >> 1;      // wave-col: 4 x 64 cols
    const int half = lane >> 4;     // 0 | 1
    const int l16  = lane & 15;

    const int m0  = blockIdx.x * BM;
    const int n0  = blockIdx.y * BN;
    const int m0w = wm * 64;
    const int n0w = wn * 64;

    // B-slice start parity is loop-invariant per lane: (kc + half*16 - n + N) & 1 == l16 & 1
    const unsigned int sh  = (unsigned int)(l16 & 1) * 16u;
    const int          odd = l16 & 1;

    // ---- one-time: build duplicated split-bf16 copy of w in LDS ----
    for (int j = tid; j < WLEN; j += 256) {
        int src = j;
        if (src >= NDIM) src -= NDIM;
        if (src >= NDIM) src -= NDIM;   // j < 2N+24 < 3N
        float wv = w[src];
        unsigned int u = __float_as_uint(wv);
        sWhi[j] = (unsigned short)(u >> 16);                       // truncation split
        float r = wv - __uint_as_float(u & 0xffff0000u);           // exact residual
        sWlo[j] = (unsigned short)(__float_as_uint(r) >> 16);
    }

    v8f acc[MF][NF];
    #pragma unroll
    for (int mf = 0; mf < MF; ++mf)
        #pragma unroll
        for (int nf = 0; nf < NF; ++nf)
            #pragma unroll
            for (int i = 0; i < 8; ++i)
                acc[mf][nf][i] = 0.0f;

    for (int kc = 0; kc < NDIM; kc += BK) {
        // ---- stage X[BM x BK] as split bf16 into LDS (perm-packed truncation) ----
        {
            const int row = tid >> 1;
            const int kp  = (tid & 1) * 16;
            const float* src = x + (size_t)(m0 + row) * NDIM + kc + kp;
            unsigned int hp[8], lp[8];
            #pragma unroll
            for (int q = 0; q < 4; ++q) {
                float4 f = ((const float4*)src)[q];
                unsigned int u0 = __float_as_uint(f.x), u1 = __float_as_uint(f.y);
                unsigned int u2 = __float_as_uint(f.z), u3 = __float_as_uint(f.w);
                hp[q * 2 + 0] = __builtin_amdgcn_perm(u1, u0, PERM_HI16);
                hp[q * 2 + 1] = __builtin_amdgcn_perm(u3, u2, PERM_HI16);
                float r0 = f.x - __uint_as_float(u0 & 0xffff0000u);  // exact residuals
                float r1 = f.y - __uint_as_float(u1 & 0xffff0000u);
                float r2 = f.z - __uint_as_float(u2 & 0xffff0000u);
                float r3 = f.w - __uint_as_float(u3 & 0xffff0000u);
                lp[q * 2 + 0] = __builtin_amdgcn_perm(__float_as_uint(r1), __float_as_uint(r0), PERM_HI16);
                lp[q * 2 + 1] = __builtin_amdgcn_perm(__float_as_uint(r3), __float_as_uint(r2), PERM_HI16);
            }
            if (kc + BK < NDIM) __builtin_prefetch(src + BK, 0, 1);
            uint4 th0{hp[0], hp[1], hp[2], hp[3]}, th1{hp[4], hp[5], hp[6], hp[7]};
            uint4 tl0{lp[0], lp[1], lp[2], lp[3]}, tl1{lp[4], lp[5], lp[6], lp[7]};
            *(uint4*)&sXhi[row * XP + kp]     = th0;
            *(uint4*)&sXhi[row * XP + kp + 8] = th1;
            *(uint4*)&sXlo[row * XP + kp]     = tl0;
            *(uint4*)&sXlo[row * XP + kp + 8] = tl1;
        }
        __syncthreads();

        // ---- A fragments (16-bit A 16x32 layout: two aligned b128 per frag) ----
        Frag32 ahi[MF], alo[MF];
        #pragma unroll
        for (int mf = 0; mf < MF; ++mf) {
            const int rb = (m0w + mf * 16 + l16) * XP + half * 16;
            ahi[mf].h[0] = *(const ushort8*)&sXhi[rb];
            ahi[mf].h[1] = *(const ushort8*)&sXhi[rb + 8];
            alo[mf].h[0] = *(const ushort8*)&sXlo[rb];
            alo[mf].h[1] = *(const ushort8*)&sXlo[rb + 8];
        }

        // ---- all B fragments upfront (no WAR on B regs during WMMA block) ----
        Frag32 bhi[NF], blo[NF];
        #pragma unroll
        for (int nf = 0; nf < NF; ++nf) {
            const int n  = n0 + n0w + nf * 16 + l16;        // output column
            const int s  = kc + half * 16 - n + NDIM;       // w-index of K-slice start (>=1)
            const int j0 = (s - odd) >> 1;                  // dword-aligned base
            const unsigned int* ph = (const unsigned int*)sWhi;
            const unsigned int* pl = (const unsigned int*)sWlo;
            unsigned int uh[9], ul[9];
            #pragma unroll
            for (int i = 0; i < 9; ++i) { uh[i] = ph[j0 + i]; ul[i] = pl[j0 + i]; }
            #pragma unroll
            for (int i = 0; i < 8; ++i) {
                bhi[nf].u[i] = ALIGNBIT(uh[i + 1], uh[i], sh);   // sh==0 -> uh[i]
                blo[nf].u[i] = ALIGNBIT(ul[i + 1], ul[i], sh);
            }
        }

        // ---- dense WMMA block: 16 independent acc chains, spacing 16 ----
        #pragma unroll
        for (int nf = 0; nf < NF; ++nf)
            #pragma unroll
            for (int mf = 0; mf < MF; ++mf)
                acc[mf][nf] = __builtin_amdgcn_wmma_f32_16x16x32_bf16(
                    false, alo[mf].v, false, bhi[nf].v, (short)0, acc[mf][nf], false, false);
        #pragma unroll
        for (int nf = 0; nf < NF; ++nf)
            #pragma unroll
            for (int mf = 0; mf < MF; ++mf)
                acc[mf][nf] = __builtin_amdgcn_wmma_f32_16x16x32_bf16(
                    false, ahi[mf].v, false, blo[nf].v, (short)0, acc[mf][nf], false, false);
        #pragma unroll
        for (int nf = 0; nf < NF; ++nf)
            #pragma unroll
            for (int mf = 0; mf < MF; ++mf)
                acc[mf][nf] = __builtin_amdgcn_wmma_f32_16x16x32_bf16(
                    false, ahi[mf].v, false, bhi[nf].v, (short)0, acc[mf][nf], false, false);
        __syncthreads();
    }

    // ---- epilogue: 16x16 f32 C/D layout -> row-major out ----
    #pragma unroll
    for (int mf = 0; mf < MF; ++mf) {
        const int mrow = m0 + m0w + mf * 16 + half * 8;
        #pragma unroll
        for (int nf = 0; nf < NF; ++nf) {
            const int nn = n0 + n0w + nf * 16 + l16;
            float* o = out + (size_t)mrow * NDIM + nn;
            #pragma unroll
            for (int v = 0; v < 8; ++v)
                o[(size_t)v * NDIM] = acc[mf][nf][v];
        }
    }
}

extern "C" void kernel_launch(void* const* d_in, const int* in_sizes, int n_in,
                              void* d_out, int out_size, void* d_ws, size_t ws_size,
                              hipStream_t stream) {
    const float* x = (const float*)d_in[0];   // [B, N] fp32
    const float* w = (const float*)d_in[1];   // [N]    fp32
    float* out = (float*)d_out;               // [B, N] fp32
    dim3 grid(BDIM / BM, NDIM / BN);          // 64 x 16 workgroups
    circulant_wmma<<<grid, 256, 0, stream>>>(x, w, out);
}
